// Net4_90486370992274
// MI455X (gfx1250) — matrix operand order repaired and checked by reading, not compiled, not run on hardware
//
#include <hip/hip_runtime.h>
#include <hip/hip_bf16.h>

// ---------------------------------------------------------------------------
// 4-layer GAT for MI455X (gfx1250, wave32).
// Dense projections: v_wmma_f32_16x16x32_bf16, A/B kept in bf16 in memory,
// f32 accumulate.  Edge softmax/aggregation: 3-pass segment softmax with
// L2-resident atomics (feature matrix = 51.2MB << 192MB L2).
// ---------------------------------------------------------------------------

typedef __attribute__((ext_vector_type(16))) __bf16 v16bf;
typedef __attribute__((ext_vector_type(8)))  __bf16 v8bf;
typedef __attribute__((ext_vector_type(8)))  float  v8f;

#define HEADS 4
#define DH    32
#define F     128      // HEADS*DH
#define SLOPE 0.2f

static __device__ __forceinline__ __bf16 f2bf(float f) { return (__bf16)f; }

// order-preserving float <-> uint encoding (for atomicMax on floats)
static __device__ __forceinline__ unsigned f2key(float f) {
  unsigned u = __float_as_uint(f);
  return (u & 0x80000000u) ? ~u : (u | 0x80000000u);
}
static __device__ __forceinline__ float key2f(unsigned k) {
  unsigned u = (k & 0x80000000u) ? (k & 0x7FFFFFFFu) : ~k;
  return __uint_as_float(u);
}
static __device__ __forceinline__ float lrelu(float e) {
  return e > 0.f ? e : SLOPE * e;
}

// f32 -> bf16 bulk convert (layer-1 input)
__global__ void convert_bf16_kernel(const float* __restrict__ in,
                                    __bf16* __restrict__ out, int cnt) {
  int i = blockIdx.x * blockDim.x + threadIdx.x;
  if (i < cnt) out[i] = f2bf(in[i]);
}

// ---------------------------------------------------------------------------
// Repack W[128x128] (row-major, K x N) into per-wave B fragments for
// v_wmma_f32_16x16x32_bf16.  Layout: [colTile 0..7][kTile 0..3][lane 0..31]
// [16 bf16].  B element (lane L, elem e): K = kt*32 + (e>=8?16:0) + (L>=16?8:0) + (e&7),
// N = t*16 + (L&15).  Each GEMM lane then loads 32 contiguous bytes.
// ---------------------------------------------------------------------------
__global__ void repack_w_kernel(const float* __restrict__ W,
                                __bf16* __restrict__ Wp) {
  int idx = blockIdx.x * blockDim.x + threadIdx.x;   // 8*4*32*16 = 16384
  if (idx >= 8 * 4 * 32 * 16) return;
  int e  = idx & 15;
  int L  = (idx >> 4) & 31;
  int kt = (idx >> 9) & 3;
  int t  = idx >> 11;
  int K  = kt * 32 + ((e & 8) ? 16 : 0) + ((L & 16) ? 8 : 0) + (e & 7);
  int Nc = t * 16 + (L & 15);
  Wp[idx] = f2bf(W[K * F + Nc]);
}

// ---------------------------------------------------------------------------
// Hf[N,128] = Xb[N,128](bf16) @ W (bf16 WMMA, f32 accumulate).
// Block = 256 threads = 8 waves; wave t owns output col tile t (16 cols),
// block owns 16 rows.  K loop = 4 x v_wmma_f32_16x16x32_bf16.
// A fragment (16x32 bf16): lane L: M = L&15;
//   elems 0..7  -> K = kt*32 + (L>=16?8:0) + e        (16B contiguous)
//   elems 8..15 -> K = kt*32 + 16 + (L>=16?8:0) + e   (16B contiguous)
// ---------------------------------------------------------------------------
__global__ __launch_bounds__(256) void gemm_wmma_kernel(
    const __bf16* __restrict__ Xb, const __bf16* __restrict__ Wp,
    float* __restrict__ Hf, int n) {
  int lane = threadIdx.x & 31;
  int t    = threadIdx.x >> 5;          // col tile 0..7
  int row0 = blockIdx.x * 16;
  int m    = lane & 15;
  int hi   = (lane & 16) ? 1 : 0;

  int arow = row0 + m; if (arow >= n) arow = n - 1;   // clamp (N%16==0 anyway)
  const __bf16* ap = Xb + (size_t)arow * F + hi * 8;
  const v16bf*  bp = (const v16bf*)Wp + t * 4 * 32 + lane;

  v8f c = {};
  #pragma unroll
  for (int kt = 0; kt < 4; ++kt) {
    v8bf alo = *(const v8bf*)(ap + kt * 32);
    v8bf ahi = *(const v8bf*)(ap + kt * 32 + 16);
    v16bf a = __builtin_shufflevector(alo, ahi,
        0, 1, 2, 3, 4, 5, 6, 7, 8, 9, 10, 11, 12, 13, 14, 15);
    v16bf b = bp[kt * 32];
    c = __builtin_amdgcn_wmma_f32_16x16x32_bf16(
        false, a, false, b, (short)0, c, false, false);
  }
  // C/D layout: VGPR r, lanes0-15: M=r, N=lane; lanes16-31: M=r+8, N=lane-16
  // Single base address; rows are compile-time offsets (r*F*4 = r*512 bytes).
  float* outp = Hf + (size_t)(row0 + hi * 8) * F + t * 16 + (lane & 15);
  if (row0 + 16 <= n) {              // full tile: unguarded immediate-offset
    #pragma unroll
    for (int r = 0; r < 8; ++r) outp[r * F] = c[r];
  } else {                           // ragged tail tile only
    #pragma unroll
    for (int r = 0; r < 8; ++r)
      if (row0 + hi * 8 + r < n) outp[r * F] = c[r];
  }
}

// el/er[N,4]: per-(node,head) dot with attention vectors
__global__ void attn_coef_kernel(const float* __restrict__ Hf,
                                 const float* __restrict__ al,
                                 const float* __restrict__ ar,
                                 float* __restrict__ el, float* __restrict__ er,
                                 int n) {
  int idx = blockIdx.x * blockDim.x + threadIdx.x;
  if (idx >= n * HEADS) return;
  int hd = idx & 3;
  const float4* hp  = (const float4*)(Hf + (size_t)(idx >> 2) * F + hd * DH);
  const float4* alp = (const float4*)(al + hd * DH);
  const float4* arp = (const float4*)(ar + hd * DH);
  float sl = 0.f, sr = 0.f;
  #pragma unroll
  for (int d = 0; d < DH / 4; ++d) {
    float4 v = hp[d], a = alp[d], r = arp[d];
    sl += v.x * a.x + v.y * a.y + v.z * a.z + v.w * a.w;
    sr += v.x * r.x + v.y * r.y + v.z * r.z + v.w * r.w;
  }
  el[idx] = sl;
  er[idx] = sr;
}

// pass 1: segment max via ordered-uint atomicMax (vector gathers)
__global__ void edge_max_kernel(const float* __restrict__ el,
                                const float* __restrict__ er,
                                const int* __restrict__ src,
                                const int* __restrict__ dst,
                                unsigned* __restrict__ mkey, int ne) {
  int i = blockIdx.x * blockDim.x + threadIdx.x;
  if (i >= ne) return;
  int s = src[i], d = dst[i];
  float4 eL = ((const float4*)el)[s];
  float4 eR = ((const float4*)er)[d];
  atomicMax(&mkey[d * 4 + 0], f2key(lrelu(eL.x + eR.x)));
  atomicMax(&mkey[d * 4 + 1], f2key(lrelu(eL.y + eR.y)));
  atomicMax(&mkey[d * 4 + 2], f2key(lrelu(eL.z + eR.z)));
  atomicMax(&mkey[d * 4 + 3], f2key(lrelu(eL.w + eR.w)));
}

// pass 2: ex = exp(e - m[dst]); store per-edge float4, segment-sum into s[dst]
__global__ void edge_expsum_kernel(const float* __restrict__ el,
                                   const float* __restrict__ er,
                                   const int* __restrict__ src,
                                   const int* __restrict__ dst,
                                   const unsigned* __restrict__ mkey,
                                   float* __restrict__ ssum,
                                   float* __restrict__ ex, int ne) {
  int i = blockIdx.x * blockDim.x + threadIdx.x;
  if (i >= ne) return;
  int s = src[i], d = dst[i];
  float4 eL = ((const float4*)el)[s];
  float4 eR = ((const float4*)er)[d];
  uint4  mk = ((const uint4*)mkey)[d];
  float4 v;
  v.x = __expf(lrelu(eL.x + eR.x) - key2f(mk.x));
  v.y = __expf(lrelu(eL.y + eR.y) - key2f(mk.y));
  v.z = __expf(lrelu(eL.z + eR.z) - key2f(mk.z));
  v.w = __expf(lrelu(eL.w + eR.w) - key2f(mk.w));
  ((float4*)ex)[i] = v;
  atomicAdd(&ssum[d * 4 + 0], v.x);
  atomicAdd(&ssum[d * 4 + 1], v.y);
  atomicAdd(&ssum[d * 4 + 2], v.z);
  atomicAdd(&ssum[d * 4 + 3], v.w);
}

// pass 3: acc[dst] += h[src] * a.  One wave32 per edge; each lane moves a
// float4 (32 lanes * 16B = full 512B feature row), 4 f32 atomics per lane.
__global__ __launch_bounds__(256) void edge_aggr_kernel(
    const float* __restrict__ Hf, const float* __restrict__ ex,
    const float* __restrict__ ssum, const int* __restrict__ src,
    const int* __restrict__ dst, float* __restrict__ acc, int ne) {
  int wv = blockIdx.x * 8 + (threadIdx.x >> 5);
  if (wv >= ne) return;
  int lane = threadIdx.x & 31;
  int s = src[wv], d = dst[wv];
  int hd = lane >> 3;                                  // 8 lanes per head
  float a = ex[(size_t)wv * 4 + hd] / ssum[d * 4 + hd];
  const float4 hv = *(const float4*)(Hf + (size_t)s * F + lane * 4);
  float* op = acc + (size_t)d * F + lane * 4;
  atomicAdd(op + 0, hv.x * a);
  atomicAdd(op + 1, hv.y * a);
  atomicAdd(op + 2, hv.z * a);
  atomicAdd(op + 3, hv.w * a);
}

// out = elu(acc + residual(x) + b); writes f32 (residual/l4) + bf16 (next GEMM)
__global__ void finalize_kernel(const float* __restrict__ acc,
                                const float* __restrict__ xin,
                                const float* __restrict__ b,
                                float* __restrict__ outp,
                                __bf16* __restrict__ outbf, int n) {
  int i = blockIdx.x * blockDim.x + threadIdx.x;
  if (i >= n * F) return;
  float v = acc[i] + xin[i] + b[i & (F - 1)];
  v = v > 0.f ? v : (__expf(v) - 1.f);
  outp[i]  = v;
  outbf[i] = f2bf(v);
}

// ----------------------- layer 4 (heads=1, out=2) --------------------------
__global__ void l4_node_kernel(const float* __restrict__ h3,
                               const float* __restrict__ W4,
                               const float* __restrict__ Wres4,
                               const float* __restrict__ al4,
                               const float* __restrict__ ar4,
                               float* __restrict__ z, float* __restrict__ res4,
                               float* __restrict__ el4, float* __restrict__ er4,
                               int n) {
  int i = blockIdx.x * blockDim.x + threadIdx.x;
  if (i >= n) return;
  const float4* hp = (const float4*)(h3 + (size_t)i * F);
  float z0 = 0.f, z1 = 0.f, r0 = 0.f, r1 = 0.f;
  #pragma unroll 8
  for (int k4 = 0; k4 < F / 4; ++k4) {
    float4 v = hp[k4];
    int k = k4 * 4;
    z0 += v.x * W4[(k+0)*2]   + v.y * W4[(k+1)*2]   + v.z * W4[(k+2)*2]   + v.w * W4[(k+3)*2];
    z1 += v.x * W4[(k+0)*2+1] + v.y * W4[(k+1)*2+1] + v.z * W4[(k+2)*2+1] + v.w * W4[(k+3)*2+1];
    r0 += v.x * Wres4[(k+0)*2]   + v.y * Wres4[(k+1)*2]   + v.z * Wres4[(k+2)*2]   + v.w * Wres4[(k+3)*2];
    r1 += v.x * Wres4[(k+0)*2+1] + v.y * Wres4[(k+1)*2+1] + v.z * Wres4[(k+2)*2+1] + v.w * Wres4[(k+3)*2+1];
  }
  z[i * 2 + 0] = z0;  z[i * 2 + 1] = z1;
  res4[i * 2 + 0] = r0;  res4[i * 2 + 1] = r1;
  el4[i] = z0 * al4[0] + z1 * al4[1];
  er4[i] = z0 * ar4[0] + z1 * ar4[1];
}

__global__ void l4_edge_max_kernel(const float* __restrict__ el4,
                                   const float* __restrict__ er4,
                                   const int* __restrict__ src,
                                   const int* __restrict__ dst,
                                   unsigned* __restrict__ mkey, int ne) {
  int i = blockIdx.x * blockDim.x + threadIdx.x;
  if (i >= ne) return;
  int d = dst[i];
  atomicMax(&mkey[d], f2key(lrelu(el4[src[i]] + er4[d])));
}

__global__ void l4_edge_expsum_kernel(const float* __restrict__ el4,
                                      const float* __restrict__ er4,
                                      const int* __restrict__ src,
                                      const int* __restrict__ dst,
                                      const unsigned* __restrict__ mkey,
                                      float* __restrict__ s4,
                                      float* __restrict__ ex4, int ne) {
  int i = blockIdx.x * blockDim.x + threadIdx.x;
  if (i >= ne) return;
  int d = dst[i];
  float v = __expf(lrelu(el4[src[i]] + er4[d]) - key2f(mkey[d]));
  ex4[i] = v;
  atomicAdd(&s4[d], v);
}

__global__ void l4_edge_aggr_kernel(const float* __restrict__ z,
                                    const float* __restrict__ ex4,
                                    const float* __restrict__ s4,
                                    const int* __restrict__ src,
                                    const int* __restrict__ dst,
                                    float* __restrict__ acc4, int ne) {
  int i = blockIdx.x * blockDim.x + threadIdx.x;
  if (i >= ne) return;
  int s = src[i], d = dst[i];
  float a = ex4[i] / s4[d];
  atomicAdd(&acc4[d * 2 + 0], z[s * 2 + 0] * a);
  atomicAdd(&acc4[d * 2 + 1], z[s * 2 + 1] * a);
}

__global__ void l4_final_kernel(const float* __restrict__ acc4,
                                const float* __restrict__ res4,
                                const float* __restrict__ b4,
                                float* __restrict__ outp, int n) {
  int i = blockIdx.x * blockDim.x + threadIdx.x;
  if (i >= n * 2) return;
  outp[i] = acc4[i] + res4[i] + b4[i & 1];   // mean over 1 head == identity
}

// ---------------------------------------------------------------------------
extern "C" void kernel_launch(void* const* d_in, const int* in_sizes, int n_in,
                              void* d_out, int out_size, void* d_ws, size_t ws_size,
                              hipStream_t stream) {
  const int N = in_sizes[0] / F;
  const int E = in_sizes[1];

  const float* x    = (const float*)d_in[0];
  const int*   src  = (const int*)d_in[1];
  const int*   dst  = (const int*)d_in[2];
  const float* Wl[3]  = { (const float*)d_in[3],  (const float*)d_in[7],  (const float*)d_in[11] };
  const float* all[3] = { (const float*)d_in[4],  (const float*)d_in[8],  (const float*)d_in[12] };
  const float* arl[3] = { (const float*)d_in[5],  (const float*)d_in[9],  (const float*)d_in[13] };
  const float* bl[3]  = { (const float*)d_in[6],  (const float*)d_in[10], (const float*)d_in[14] };
  const float* W4    = (const float*)d_in[15];
  const float* al4   = (const float*)d_in[16];
  const float* ar4   = (const float*)d_in[17];
  const float* b4    = (const float*)d_in[18];
  const float* Wres4 = (const float*)d_in[19];

  // ---- workspace carving ----
  char* base = (char*)d_ws;
  const size_t NB  = (size_t)N * F * sizeof(float);     // 51.2 MB
  const size_t NBH = (size_t)N * F * sizeof(__bf16);    // 25.6 MB
  float*    big0 = (float*)(base);
  float*    big1 = (float*)(base + NB);
  float*    big2 = (float*)(base + 2 * NB);
  char* p = base + 3 * NB;
  __bf16*   bf0  = (__bf16*)p;           p += NBH;
  __bf16*   bf1  = (__bf16*)p;           p += NBH;
  float*    el   = (float*)p;            p += (size_t)N * 4 * 4;
  float*    er   = (float*)p;            p += (size_t)N * 4 * 4;
  unsigned* mkey = (unsigned*)p;         p += (size_t)N * 4 * 4;
  float*    sbuf = (float*)p;            p += (size_t)N * 4 * 4;
  float*    ex   = (float*)p;            p += (size_t)E * 4 * 4;   // 25.6 MB
  __bf16*   Wp   = (__bf16*)p;           p += 8 * 4 * 32 * 16 * 2;
  float*    z    = (float*)p;            p += (size_t)N * 2 * 4;
  float*    res4 = (float*)p;            p += (size_t)N * 2 * 4;
  float*    el4  = (float*)p;            p += (size_t)N * 4;
  float*    er4  = (float*)p;            p += (size_t)N * 4;
  unsigned* mk4  = (unsigned*)p;         p += (size_t)N * 4;
  float*    s4   = (float*)p;            p += (size_t)N * 4;
  float*    acc4 = (float*)p;            p += (size_t)N * 2 * 4;
  float*    ex4  = ex;                   // reuse (layers 1-3 done by layer 4)

  const int TPB = 256;
  const int gN4  = (N * 4 + TPB - 1) / TPB;
  const int gNF  = (N * F + TPB - 1) / TPB;
  const int gE   = (E + TPB - 1) / TPB;
  const int gEw  = (E + 7) / 8;            // one wave32 per edge
  const int gRow = (N + 15) / 16;

  // bf16 A-operand rotation: xbf=bf0 -> L1 out bf1 -> L2 out bf0 -> L3 out bf1
  convert_bf16_kernel<<<gNF, TPB, 0, stream>>>(x, bf0, N * F);

  // f32 buffer rotation: (X, Hfeat, Acc, Out)
  const float* Xs[3]   = { x,    big2, big0 };
  float*       Hfs[3]  = { big0, big0, big1 };
  float*       Acs[3]  = { big1, big1, big2 };
  float*       Outs[3] = { big2, big0, big1 };
  __bf16*      Ain[3]  = { bf0,  bf1,  bf0  };
  __bf16*      Abf[3]  = { bf1,  bf0,  bf1  };

  for (int li = 0; li < 3; ++li) {
    const float* X  = Xs[li];
    float* Hf  = Hfs[li];
    float* Acc = Acs[li];
    float* Out = Outs[li];

    hipMemsetAsync(Acc,  0, NB, stream);
    hipMemsetAsync(mkey, 0, (size_t)N * 4 * sizeof(unsigned), stream);
    hipMemsetAsync(sbuf, 0, (size_t)N * 4 * sizeof(float), stream);

    repack_w_kernel<<<64, TPB, 0, stream>>>(Wl[li], Wp);
    gemm_wmma_kernel<<<gRow, TPB, 0, stream>>>(Ain[li], Wp, Hf, N);
    attn_coef_kernel<<<gN4, TPB, 0, stream>>>(Hf, all[li], arl[li], el, er, N);
    edge_max_kernel<<<gE, TPB, 0, stream>>>(el, er, src, dst, mkey, E);
    edge_expsum_kernel<<<gE, TPB, 0, stream>>>(el, er, src, dst, mkey, sbuf, ex, E);
    edge_aggr_kernel<<<gEw, TPB, 0, stream>>>(Hf, ex, sbuf, src, dst, Acc, E);
    finalize_kernel<<<gNF, TPB, 0, stream>>>(Acc, X, bl[li], Out, Abf[li], N);
  }

  // ---- layer 4: in=128 -> heads=1, out=2, projected residual ----
  const float* h3 = big1;
  hipMemsetAsync(acc4, 0, (size_t)N * 2 * sizeof(float), stream);
  hipMemsetAsync(mk4,  0, (size_t)N * sizeof(unsigned), stream);
  hipMemsetAsync(s4,   0, (size_t)N * sizeof(float), stream);

  l4_node_kernel<<<(N + TPB - 1) / TPB, TPB, 0, stream>>>(
      h3, W4, Wres4, al4, ar4, z, res4, el4, er4, N);
  l4_edge_max_kernel<<<gE, TPB, 0, stream>>>(el4, er4, src, dst, mk4, E);
  l4_edge_expsum_kernel<<<gE, TPB, 0, stream>>>(el4, er4, src, dst, mk4, s4, ex4, E);
  l4_edge_aggr_kernel<<<gE, TPB, 0, stream>>>(z, ex4, s4, src, dst, acc4, E);
  l4_final_kernel<<<(N * 2 + TPB - 1) / TPB, TPB, 0, stream>>>(
      acc4, res4, b4, (float*)d_out, N);
}
